// PointNetEncoder_63007170232691
// MI455X (gfx1250) — matrix operand order repaired and checked by reading, not compiled, not run on hardware
//
#include <hip/hip_runtime.h>
#include <math.h>

namespace {

constexpr int   B       = 16;
constexpr int   NPTS    = 2048;
constexpr int   KNN     = 20;
constexpr float EPS     = 1e-6f;
constexpr float BN_EPS  = 1e-5f;
constexpr float NEG_BIG = -3.0e38f;

typedef __attribute__((ext_vector_type(16))) _Float16 v16h;
typedef __attribute__((ext_vector_type(8)))  _Float16 v8h;
typedef __attribute__((ext_vector_type(8)))  float    v8f;

// ---------------------------------------------------------------------------
// 1. kNN top-K (by negative squared distance, self included, ties -> low idx)
// ---------------------------------------------------------------------------
__global__ void __launch_bounds__(256)
knn_topk_kernel(const float* __restrict__ x, int* __restrict__ idx) {
  __shared__ float dist[NPTS];
  __shared__ float sv[256];
  __shared__ int   si[256];
  const int p = blockIdx.x;
  const int b = p / NPTS;
  const int n = p % NPTS;
  const float* xb = x + (size_t)b * 3 * NPTS;
  const float cx = xb[n], cy = xb[NPTS + n], cz = xb[2 * NPTS + n];
  const float cs = cx * cx + cy * cy + cz * cz;
  for (int m = threadIdx.x; m < NPTS; m += 256) {
    float px = xb[m], py = xb[NPTS + m], pz = xb[2 * NPTS + m];
    dist[m] = 2.f * (cx * px + cy * py + cz * pz) - cs -
              (px * px + py * py + pz * pz);
  }
  __syncthreads();
  for (int it = 0; it < KNN; ++it) {
    float bv = NEG_BIG;
    int   bi = 0x7fffffff;
    for (int m = threadIdx.x; m < NPTS; m += 256)
      if (dist[m] > bv) { bv = dist[m]; bi = m; }
    sv[threadIdx.x] = bv;
    si[threadIdx.x] = bi;
    __syncthreads();
    for (int o = 128; o > 0; o >>= 1) {
      if (threadIdx.x < o) {
        float ov = sv[threadIdx.x + o];
        int   oi = si[threadIdx.x + o];
        if (ov > sv[threadIdx.x] ||
            (ov == sv[threadIdx.x] && oi < si[threadIdx.x])) {
          sv[threadIdx.x] = ov;
          si[threadIdx.x] = oi;
        }
      }
      __syncthreads();
    }
    if (threadIdx.x == 0) {
      idx[(size_t)p * KNN + it] = si[0];
      dist[si[0]] = NEG_BIG;
    }
    __syncthreads();
  }
}

// ---------------------------------------------------------------------------
// 2. Edge features: feat[B, 3(chan), 3(vec), N, K] = {nb-ctr, ctr, nb x ctr}
// ---------------------------------------------------------------------------
__global__ void __launch_bounds__(256)
edge_feat_kernel(const float* __restrict__ x, const int* __restrict__ idx,
                 float* __restrict__ feat) {
  const int t = blockIdx.x * 256 + threadIdx.x;
  const int total = B * NPTS * KNN;
  if (t >= total) return;
  const int k = t % KNN;
  const int n = (t / KNN) % NPTS;
  const int b = t / (KNN * NPTS);
  const int i = idx[t];
  const float cx = x[(b * 3 + 0) * NPTS + n];
  const float cy = x[(b * 3 + 1) * NPTS + n];
  const float cz = x[(b * 3 + 2) * NPTS + n];
  const float nx = x[(b * 3 + 0) * NPTS + i];
  const float ny = x[(b * 3 + 1) * NPTS + i];
  const float nz = x[(b * 3 + 2) * NPTS + i];
  const float d0 = nx - cx, d1 = ny - cy, d2 = nz - cz;
  const float r0 = ny * cz - nz * cy;
  const float r1 = nz * cx - nx * cz;
  const float r2 = nx * cy - ny * cx;
  const size_t s  = (size_t)n * KNN + k;
  const size_t cs = (size_t)NPTS * KNN;
  float* fb = feat + (size_t)b * 9 * cs;
  fb[0 * cs + s] = d0; fb[1 * cs + s] = d1; fb[2 * cs + s] = d2;
  fb[3 * cs + s] = cx; fb[4 * cs + s] = cy; fb[5 * cs + s] = cz;
  fb[6 * cs + s] = r0; fb[7 * cs + s] = r1; fb[8 * cs + s] = r2;
}

// ---------------------------------------------------------------------------
// 3a. Fast WMMA GEMM, requires S3 % 128 == 0.
//     Per batch: Y[M x S3] = W[M x Kd] * X[Kd x S3].
//     Block = 256 thr (8 waves) -> 32(M) x 128(N) tile.
//     Staging is 8-wide: 8 batched global_load_b32 -> 8 cvt -> 1 ds_store_b128.
//     Inner loop: 16B LDS fragment loads + 2 WMMA/wave, branchless.
// ---------------------------------------------------------------------------
__global__ void __launch_bounds__(256)
vn_gemm_wmma_fast(const float* __restrict__ W, const float* __restrict__ X,
                  float* __restrict__ Y, int M, int Kd, int S3) {
  extern __shared__ __align__(16) _Float16 smem[];
  const int Kp = (Kd + 31) & ~31;
  _Float16* wl = smem;              // [32][Kp]
  _Float16* xl = smem + 32 * Kp;    // [128][32]  (transposed panel: [n][k])

  const int tid  = threadIdx.x;
  const int lane = tid & 31;
  const int wave = tid >> 5;
  const int b    = blockIdx.z;
  const int mblk = blockIdx.y * 32;
  const int nblk = blockIdx.x * 128;
  const float* Xb = X + (size_t)b * Kd * S3;
  float*       Yb = Y + (size_t)b * M * S3;

  // ---- stage W tile (one-time): item = (row, 8-wide K-run) ----
  {
    const int kocts = Kp >> 3;               // K-runs per row
    for (int id = tid; id < 32 * kocts; id += 256) {
      const int r  = id / kocts;
      const int k8 = (id - r * kocts) * 8;
      const int m  = mblk + r;
      const int mc = m < M ? m : (M - 1);
      float f[8];
#pragma unroll
      for (int j = 0; j < 8; ++j) {
        const int kk = k8 + j;
        const int kc = kk < Kd ? kk : (Kd - 1);
        f[j] = W[(size_t)mc * Kd + kc];
      }
      v8h h;
#pragma unroll
      for (int j = 0; j < 8; ++j) {
        const int kk = k8 + j;
        h[j] = (m < M && kk < Kd) ? (_Float16)f[j] : (_Float16)0.f;
      }
      *(v8h*)(wl + (size_t)r * Kp + k8) = h;
    }
  }

  const int mw   = wave >> 2;                 // 0..1
  const int nw   = wave & 3;                  // 0..3
  const int lrow = mw * 16 + (lane & 15);     // local W row
  const int hi   = (lane >> 4) & 1;
  const int nloc = nw * 32 + (lane & 15);     // local col of acc0
  const int ncol = nblk + nloc;               // global col of acc0

  // staging coords for X panels: item = (koct 0..3, nn 0..127) x2 per thread
  const int snn0 = tid & 127;
  const int sk0  = (tid >> 7) * 8;            // 0 or 8
  v8f acc0 = {};
  v8f acc1 = {};

  for (int k0 = 0; k0 < Kd; k0 += 32) {
    __syncthreads();
    // ---- stage X panel transposed, branchless, 8-wide ----
#pragma unroll
    for (int half = 0; half < 2; ++half) {
      const int nn = snn0;
      const int k8 = sk0 + half * 16;         // 0,16 or 8,24
      float f[8];
#pragma unroll
      for (int j = 0; j < 8; ++j) {
        const int k  = k0 + k8 + j;
        const int kc = k < Kd ? k : (Kd - 1);
        f[j] = Xb[(size_t)kc * S3 + nblk + nn];
      }
      v8h h;
#pragma unroll
      for (int j = 0; j < 8; ++j) {
        const int k = k0 + k8 + j;
        h[j] = (k < Kd) ? (_Float16)f[j] : (_Float16)0.f;
      }
      *(v8h*)(xl + (size_t)nn * 32 + k8) = h;
      // branchless prefetch of next panel (clamped, speculative)
      const int kn = (k0 + 32 + k8) < Kd ? (k0 + 32 + k8) : (Kd - 1);
      __builtin_prefetch(&Xb[(size_t)kn * S3 + nblk + nn], 0, 1);
    }
    __syncthreads();

    // ---- fragments from LDS (16B aligned vector loads) ----
    const int akb = k0 + hi * 16;
    const v8h a0 = *(const v8h*)(wl + (size_t)lrow * Kp + akb);
    const v8h a1 = *(const v8h*)(wl + (size_t)lrow * Kp + akb + 8);
    const v16h a = __builtin_shufflevector(a0, a1, 0, 1, 2, 3, 4, 5, 6, 7,
                                           8, 9, 10, 11, 12, 13, 14, 15);
    const int bkb = hi * 16;
    const v8h b00 = *(const v8h*)(xl + nloc * 32 + bkb);
    const v8h b01 = *(const v8h*)(xl + nloc * 32 + bkb + 8);
    const v16h bf0 = __builtin_shufflevector(b00, b01, 0, 1, 2, 3, 4, 5, 6, 7,
                                             8, 9, 10, 11, 12, 13, 14, 15);
    const v8h b10 = *(const v8h*)(xl + (nloc + 16) * 32 + bkb);
    const v8h b11 = *(const v8h*)(xl + (nloc + 16) * 32 + bkb + 8);
    const v16h bf1 = __builtin_shufflevector(b10, b11, 0, 1, 2, 3, 4, 5, 6, 7,
                                             8, 9, 10, 11, 12, 13, 14, 15);

    acc0 = __builtin_amdgcn_wmma_f32_16x16x32_f16(false, a, false, bf0,
                                                  (short)0, acc0, false, false);
    acc1 = __builtin_amdgcn_wmma_f32_16x16x32_f16(false, a, false, bf1,
                                                  (short)0, acc1, false, false);
  }

#pragma unroll
  for (int v = 0; v < 8; ++v) {
    const int m = mblk + mw * 16 + v + hi * 8;
    if (m < M) {
      Yb[(size_t)m * S3 + ncol]      = acc0[v];
      Yb[(size_t)m * S3 + ncol + 16] = acc1[v];
    }
  }
}

// ---------------------------------------------------------------------------
// 3b. Small guarded WMMA GEMM (any S3); branchless clamped loads + selects.
// ---------------------------------------------------------------------------
__global__ void __launch_bounds__(256)
vn_gemm_wmma_small(const float* __restrict__ W, const float* __restrict__ X,
                   float* __restrict__ Y, int M, int Kd, int S3) {
  const int lane = threadIdx.x & 31;
  const int wave = threadIdx.x >> 5;
  const int b    = blockIdx.z;
  const int m0   = blockIdx.y * 32 + (wave >> 2) * 16;
  const int n0   = blockIdx.x * 64 + (wave & 3) * 16;
  const float* Xb = X + (size_t)b * Kd * S3;
  float*       Yb = Y + (size_t)b * M * S3;

  v8f acc = {};
  const int  mrow = m0 + (lane & 15);
  const int  mc   = mrow < M ? mrow : (M - 1);
  const bool mok  = (mrow < M);
  const int  ncol = n0 + (lane & 15);
  const int  nc   = ncol < S3 ? ncol : (S3 - 1);
  const bool nok  = (ncol < S3);
  const int  hi   = (lane >> 4) & 1;

  for (int k0 = 0; k0 < Kd; k0 += 32) {
    v16h a, bf;
    const int akb = k0 + hi * 16;
#pragma unroll
    for (int e = 0; e < 8; ++e) {
      const int k1 = akb + e;
      const int k2 = akb + 8 + e;
      const int kc1 = k1 < Kd ? k1 : (Kd - 1);
      const int kc2 = k2 < Kd ? k2 : (Kd - 1);
      const float w1 = W[(size_t)mc * Kd + kc1];
      const float w2 = W[(size_t)mc * Kd + kc2];
      a[e]     = (mok && k1 < Kd) ? (_Float16)w1 : (_Float16)0.f;
      a[e + 8] = (mok && k2 < Kd) ? (_Float16)w2 : (_Float16)0.f;
    }
    const int bkb = k0 + hi * 16;
#pragma unroll
    for (int e = 0; e < 16; ++e) {
      const int k  = bkb + e;
      const int kc = k < Kd ? k : (Kd - 1);
      const float v = Xb[(size_t)kc * S3 + nc];
      bf[e] = (nok && k < Kd) ? (_Float16)v : (_Float16)0.f;
    }
    acc = __builtin_amdgcn_wmma_f32_16x16x32_f16(false, a, false, bf, (short)0,
                                                 acc, false, false);
  }
#pragma unroll
  for (int v = 0; v < 8; ++v) {
    const int m = m0 + v + hi * 8;
    if (m < M && nok) Yb[(size_t)m * S3 + ncol] = acc[v];
  }
}

// ---------------------------------------------------------------------------
// 4. BN training stats of per-vector norms: per channel over (B, S)
// ---------------------------------------------------------------------------
__global__ void __launch_bounds__(256)
bn_stats_kernel(const float* __restrict__ P, float* __restrict__ stats,
                int C, int S) {
  __shared__ float s1s[256], s2s[256];
  const int c = blockIdx.x;
  float s1 = 0.f, s2 = 0.f;
  const int tot = B * S;
  for (int t = threadIdx.x; t < tot; t += 256) {
    const int b = t / S;
    const int s = t - b * S;
    const float* base = P + ((size_t)(b * C + c) * 3) * S + s;
    const float x0 = base[0], x1 = base[S], x2 = base[2 * (size_t)S];
    const float n = sqrtf(x0 * x0 + x1 * x1 + x2 * x2) + EPS;
    s1 += n;
    s2 += n * n;
  }
  s1s[threadIdx.x] = s1;
  s2s[threadIdx.x] = s2;
  __syncthreads();
  for (int o = 128; o > 0; o >>= 1) {
    if (threadIdx.x < o) {
      s1s[threadIdx.x] += s1s[threadIdx.x + o];
      s2s[threadIdx.x] += s2s[threadIdx.x + o];
    }
    __syncthreads();
  }
  if (threadIdx.x == 0) {
    const float inv = 1.f / (float)tot;
    const float mu  = s1s[0] * inv;
    stats[2 * c]     = mu;
    stats[2 * c + 1] = s2s[0] * inv - mu * mu;
  }
}

// ---------------------------------------------------------------------------
// 5. BN rescale (+ optional projection leaky-ReLU, slope 0), in place on P
// ---------------------------------------------------------------------------
__global__ void __launch_bounds__(256)
llr_apply_kernel(float* __restrict__ P, const float* __restrict__ D,
                 const float* __restrict__ stats, const float* __restrict__ g,
                 const float* __restrict__ bb, int C, int S, int total,
                 int do_lrelu) {
  const int t = blockIdx.x * 256 + threadIdx.x;
  if (t >= total) return;
  const int s = t % S;
  const int c = (t / S) % C;
  const int b = t / (S * C);
  const size_t base = ((size_t)(b * C + c) * 3) * S + s;
  float p0 = P[base], p1 = P[base + S], p2 = P[base + 2 * (size_t)S];
  const float n   = sqrtf(p0 * p0 + p1 * p1 + p2 * p2) + EPS;
  const float mu  = stats[2 * c];
  const float var = stats[2 * c + 1];
  const float nbn = (n - mu) * rsqrtf(var + BN_EPS) * g[c] + bb[c];
  const float sc  = nbn / n;
  p0 *= sc; p1 *= sc; p2 *= sc;
  if (do_lrelu) {
    const float d0 = D[base], d1 = D[base + S], d2 = D[base + 2 * (size_t)S];
    const float dot = p0 * d0 + p1 * d1 + p2 * d2;
    if (dot < 0.f) {
      const float f = dot / (d0 * d0 + d1 * d1 + d2 * d2 + EPS);
      p0 -= f * d0; p1 -= f * d1; p2 -= f * d2;
    }
  }
  P[base] = p0; P[base + S] = p1; P[base + 2 * (size_t)S] = p2;
}

// ---------------------------------------------------------------------------
// 6. VN max-pool over last axis L: argmax of <x, d>, gather x
// ---------------------------------------------------------------------------
__global__ void __launch_bounds__(256)
vn_pool_kernel(const float* __restrict__ X, const float* __restrict__ D,
               float* __restrict__ Y, int C, int SO, int L, int total) {
  const int t = blockIdx.x * 256 + threadIdx.x;
  if (t >= total) return;
  const int so = t % SO;
  const int c  = (t / SO) % C;
  const int b  = t / (SO * C);
  const size_t SOL  = (size_t)SO * L;
  const size_t base = ((size_t)(b * C + c) * 3) * SOL + (size_t)so * L;
  float best = NEG_BIG;
  int   bl   = 0;
  for (int l = 0; l < L; ++l) {
    const float dot = X[base + l] * D[base + l] +
                      X[base + SOL + l] * D[base + SOL + l] +
                      X[base + 2 * SOL + l] * D[base + 2 * SOL + l];
    if (dot > best) { best = dot; bl = l; }
  }
  const size_t ob = ((size_t)(b * C + c) * 3) * SO + so;
  Y[ob]          = X[base + bl];
  Y[ob + SO]     = X[base + SOL + bl];
  Y[ob + 2 * SO] = X[base + 2 * SOL + bl];
}

// ---------------------------------------------------------------------------
// 7. Mean over last axis (one thread per row)
// ---------------------------------------------------------------------------
__global__ void __launch_bounds__(256)
mean_lastaxis_kernel(const float* __restrict__ X, float* __restrict__ Y,
                     int rows, int L) {
  const int t = blockIdx.x * 256 + threadIdx.x;
  if (t >= rows) return;
  const float* xr = X + (size_t)t * L;
  float s = 0.f;
  for (int l = 0; l < L; ++l) s += xr[l];
  Y[t] = s / (float)L;
}

// ---------------------------------------------------------------------------
// 8. Channel concat: dst[B,Cm+Cb,3,S] = [main | broadcast-over-S]
// ---------------------------------------------------------------------------
__global__ void __launch_bounds__(256)
concat_bcast_kernel(float* __restrict__ dst, const float* __restrict__ mainp,
                    const float* __restrict__ bc, int Cm, int Cb, int S,
                    int total) {
  const int t = blockIdx.x * 256 + threadIdx.x;
  if (t >= total) return;
  const int Ct = Cm + Cb;
  const int s  = t % S;
  const int dd = (t / S) % 3;
  const int c  = (t / (S * 3)) % Ct;
  const int b  = t / (S * 3 * Ct);
  float v;
  if (c < Cm)
    v = mainp[((size_t)(b * Cm + c) * 3 + dd) * S + s];
  else
    v = bc[(size_t)(b * Cb + (c - Cm)) * 3 + dd];
  dst[t] = v;
}

// ---------------------------------------------------------------------------
// 9. Final: out[b, i*3+k] = max_n sum_j h[b,i,j,n] * z0[b,j,k,n]
// ---------------------------------------------------------------------------
__global__ void __launch_bounds__(256)
final_max_kernel(const float* __restrict__ h, const float* __restrict__ z0,
                 float* __restrict__ out) {
  __shared__ float r0[256], r1[256], r2[256];
  const int b = blockIdx.x / 682;
  const int i = blockIdx.x % 682;
  const float* hb = h + ((size_t)(b * 682 + i) * 3) * NPTS;
  const float* zb = z0 + (size_t)b * 9 * NPTS;
  float m0 = NEG_BIG, m1 = NEG_BIG, m2 = NEG_BIG;
  for (int n = threadIdx.x; n < NPTS; n += 256) {
    const float h0 = hb[n], h1 = hb[NPTS + n], h2 = hb[2 * NPTS + n];
    m0 = fmaxf(m0, h0 * zb[0 * NPTS + n] + h1 * zb[3 * NPTS + n] +
                       h2 * zb[6 * NPTS + n]);
    m1 = fmaxf(m1, h0 * zb[1 * NPTS + n] + h1 * zb[4 * NPTS + n] +
                       h2 * zb[7 * NPTS + n]);
    m2 = fmaxf(m2, h0 * zb[2 * NPTS + n] + h1 * zb[5 * NPTS + n] +
                       h2 * zb[8 * NPTS + n]);
  }
  r0[threadIdx.x] = m0; r1[threadIdx.x] = m1; r2[threadIdx.x] = m2;
  __syncthreads();
  for (int o = 128; o > 0; o >>= 1) {
    if (threadIdx.x < o) {
      r0[threadIdx.x] = fmaxf(r0[threadIdx.x], r0[threadIdx.x + o]);
      r1[threadIdx.x] = fmaxf(r1[threadIdx.x], r1[threadIdx.x + o]);
      r2[threadIdx.x] = fmaxf(r2[threadIdx.x], r2[threadIdx.x + o]);
    }
    __syncthreads();
  }
  if (threadIdx.x == 0) {
    float* ob = out + (size_t)b * 2046 + (size_t)i * 3;
    ob[0] = r0[0]; ob[1] = r1[0]; ob[2] = r2[0];
  }
}

}  // namespace

// ---------------------------------------------------------------------------
extern "C" void kernel_launch(void* const* d_in, const int* in_sizes, int n_in,
                              void* d_out, int out_size, void* d_ws,
                              size_t ws_size, hipStream_t stream) {
  (void)in_sizes; (void)n_in; (void)out_size; (void)ws_size;
  const float* x      = (const float*)d_in[0];
  const float* cp_wf  = (const float*)d_in[1];
  const float* cp_wd  = (const float*)d_in[2];
  const float* cp_g   = (const float*)d_in[3];
  const float* cp_b   = (const float*)d_in[4];
  const float* c1_wf  = (const float*)d_in[5];
  const float* c1_wd  = (const float*)d_in[6];
  const float* c1_g   = (const float*)d_in[7];
  const float* c1_b   = (const float*)d_in[8];
  const float* pool_wd= (const float*)d_in[9];
  const float* s1_wf  = (const float*)d_in[10];
  const float* s1_wd  = (const float*)d_in[11];
  const float* s1_g   = (const float*)d_in[12];
  const float* s1_b   = (const float*)d_in[13];
  const float* s2_wf  = (const float*)d_in[14];
  const float* s2_wd  = (const float*)d_in[15];
  const float* s2_g   = (const float*)d_in[16];
  const float* s2_b   = (const float*)d_in[17];
  const float* s3_wf  = (const float*)d_in[18];
  const float* s3_wd  = (const float*)d_in[19];
  const float* s3_g   = (const float*)d_in[20];
  const float* s3_b   = (const float*)d_in[21];
  const float* sp_wd  = (const float*)d_in[22];
  const float* sf1_wf = (const float*)d_in[23];
  const float* sf1_wd = (const float*)d_in[24];
  const float* sf1_g  = (const float*)d_in[25];
  const float* sf1_b  = (const float*)d_in[26];
  const float* sf2_wf = (const float*)d_in[27];
  const float* sf2_wd = (const float*)d_in[28];
  const float* sf2_g  = (const float*)d_in[29];
  const float* sf2_b  = (const float*)d_in[30];
  const float* sf3_w  = (const float*)d_in[31];
  const float* c2_wf  = (const float*)d_in[32];
  const float* c2_wd  = (const float*)d_in[33];
  const float* c2_g   = (const float*)d_in[34];
  const float* c2_b   = (const float*)d_in[35];
  const float* c3_w   = (const float*)d_in[36];
  const float* bn3_g  = (const float*)d_in[37];
  const float* bn3_b  = (const float*)d_in[38];
  const float* v1_wf  = (const float*)d_in[39];
  const float* v1_wd  = (const float*)d_in[40];
  const float* v1_g   = (const float*)d_in[41];
  const float* v1_b   = (const float*)d_in[42];
  const float* v2_wf  = (const float*)d_in[43];
  const float* v2_wd  = (const float*)d_in[44];
  const float* v2_g   = (const float*)d_in[45];
  const float* v2_b   = (const float*)d_in[46];
  const float* vlin_w = (const float*)d_in[47];

  char* wsb = (char*)d_ws;
  size_t off = 0;
  auto alloc_f = [&](size_t n) -> float* {
    off = (off + 255) & ~(size_t)255;
    float* p = (float*)(wsb + off);
    off += n * 4;
    return p;
  };
  auto alloc_i = [&](size_t n) -> int* {
    off = (off + 255) & ~(size_t)255;
    int* p = (int*)(wsb + off);
    off += n * 4;
    return p;
  };

  constexpr size_t SNK3 = (size_t)3 * NPTS * KNN;  // 122880 (mult of 128)
  constexpr size_t SN3  = (size_t)3 * NPTS;        // 6144   (mult of 128)

  int*   idxb  = alloc_i((size_t)B * NPTS * KNN);
  float* feat  = alloc_f((size_t)B * 9 * NPTS * KNN);
  float* A1    = alloc_f((size_t)B * 21 * SNK3);   // >= B*341*SN3 too
  float* A2    = alloc_f((size_t)B * 21 * SNK3);
  float* A0    = alloc_f((size_t)B * 682 * SN3);   // h682
  float* h21   = alloc_f((size_t)B * 21 * SN3);
  float* h1    = alloc_f((size_t)B * 21 * SN3);
  float* t1    = alloc_f((size_t)B * 21 * SN3);
  float* t2    = alloc_f((size_t)B * 42 * SN3);
  float* h42   = alloc_f((size_t)B * 42 * SN3);
  float* t341  = alloc_f((size_t)B * 341 * 3);
  float* tsf1  = alloc_f((size_t)B * 170 * 3);
  float* tsf2  = alloc_f((size_t)B * 85 * 3);
  float* t21v  = alloc_f((size_t)B * 21 * 3);
  float* dsml  = alloc_f((size_t)B * 341 * 3);
  float* means = alloc_f((size_t)B * 341 * 3);
  float* z0    = alloc_f((size_t)B * 9 * NPTS);
  float* stats = alloc_f(2 * 682);

  auto gemm = [&](const float* Wp, const float* Xp, float* Yp, int M, int Kd,
                  int S3) {
    if ((S3 & 127) == 0) {
      const int Kp = (Kd + 31) & ~31;
      const size_t shm = (size_t)(32 * Kp + 128 * 32) * sizeof(_Float16);
      dim3 grid(S3 / 128, (M + 31) / 32, B);
      vn_gemm_wmma_fast<<<grid, 256, shm, stream>>>(Wp, Xp, Yp, M, Kd, S3);
    } else {
      dim3 grid((S3 + 63) / 64, (M + 31) / 32, B);
      vn_gemm_wmma_small<<<grid, 256, 0, stream>>>(Wp, Xp, Yp, M, Kd, S3);
    }
  };
  auto bnst = [&](const float* Pp, int C, int S) {
    bn_stats_kernel<<<C, 256, 0, stream>>>(Pp, stats, C, S);
  };
  auto apply = [&](float* Pp, const float* Dp, const float* g, const float* bb,
                   int C, int S, int lrelu) {
    const int total = B * C * S;
    llr_apply_kernel<<<(total + 255) / 256, 256, 0, stream>>>(
        Pp, Dp, stats, g, bb, C, S, total, lrelu);
  };
  auto llr = [&](const float* Xp, const float* wf, const float* wd,
                 const float* g, const float* bb, float* Pp, float* Dp,
                 int Cin, int Cout, int S) {
    gemm(wf, Xp, Pp, Cout, Cin, 3 * S);
    gemm(wd, Xp, Dp, Cout, Cin, 3 * S);
    bnst(Pp, Cout, S);
    apply(Pp, Dp, g, bb, Cout, S, 1);
  };

  // ---- pipeline ----
  knn_topk_kernel<<<B * NPTS, 256, 0, stream>>>(x, idxb);
  {
    const int total = B * NPTS * KNN;
    edge_feat_kernel<<<(total + 255) / 256, 256, 0, stream>>>(x, idxb, feat);
  }
  // cp llr on edge features (S = N*K); result -> A1
  llr(feat, cp_wf, cp_wd, cp_g, cp_b, A1, A2, 3, 21, NPTS * KNN);
  // max-pool over K -> h21
  gemm(pool_wd, A1, A2, 21, 21, (int)SNK3);
  {
    const int total = B * 21 * NPTS;
    vn_pool_kernel<<<(total + 255) / 256, 256, 0, stream>>>(A1, A2, h21, 21,
                                                            NPTS, KNN, total);
  }
  // c1 -> h1
  llr(h21, c1_wf, c1_wd, c1_g, c1_b, h1, A2, 21, 21, NPTS);
  // STN trunk
  llr(h1, s1_wf, s1_wd, s1_g, s1_b, t1, A2, 21, 21, NPTS);
  llr(t1, s2_wf, s2_wd, s2_g, s2_b, t2, A2, 21, 42, NPTS);
  llr(t2, s3_wf, s3_wd, s3_g, s3_b, A1, A2, 42, 341, NPTS);  // t3 -> A1
  // sp max-pool over N -> t341 [B,341,3]
  gemm(sp_wd, A1, A2, 341, 341, (int)SN3);
  {
    const int total = B * 341;
    vn_pool_kernel<<<(total + 255) / 256, 256, 0, stream>>>(A1, A2, t341, 341,
                                                            1, NPTS, total);
  }
  llr(t341, sf1_wf, sf1_wd, sf1_g, sf1_b, tsf1, dsml, 341, 170, 1);
  llr(tsf1, sf2_wf, sf2_wd, sf2_g, sf2_b, tsf2, dsml, 170, 85, 1);
  gemm(sf3_w, tsf2, t21v, 21, 85, 3);  // t21 [B,21,3]
  // concat(h1, broadcast t21) -> h42
  {
    const int total = B * 42 * 3 * NPTS;
    concat_bcast_kernel<<<(total + 255) / 256, 256, 0, stream>>>(
        h42, h1, t21v, 21, 21, NPTS, total);
  }
  // c2 -> A1
  llr(h42, c2_wf, c2_wd, c2_g, c2_b, A1, A2, 42, 42, NPTS);
  // c3 lin + bn (no lrelu) -> A2 (h3)
  gemm(c3_w, A1, A2, 341, 42, (int)SN3);
  bnst(A2, 341, NPTS);
  apply(A2, A2, bn3_g, bn3_b, 341, NPTS, 0);
  // mean over N, concat(h3, broadcast mean) -> A0 (h682)
  {
    const int rows = B * 341 * 3;
    mean_lastaxis_kernel<<<(rows + 255) / 256, 256, 0, stream>>>(A2, means,
                                                                 rows, NPTS);
  }
  {
    const int total = B * 682 * 3 * NPTS;
    concat_bcast_kernel<<<(total + 255) / 256, 256, 0, stream>>>(
        A0, A2, means, 341, 341, NPTS, total);
  }
  // VNStdFeature: v1 -> A1 (z1), v2 -> A2 (z2), vlin -> z0
  llr(A0, v1_wf, v1_wd, v1_g, v1_b, A1, A2, 682, 341, NPTS);
  {
    float* Pp = A2;
    float* Dp = A2 + (size_t)B * 170 * SN3;
    gemm(v2_wf, A1, Pp, 170, 341, (int)SN3);
    gemm(v2_wd, A1, Dp, 170, 341, (int)SN3);
    bnst(Pp, 170, NPTS);
    apply(Pp, Dp, v2_g, v2_b, 170, NPTS, 1);
  }
  gemm(vlin_w, A2, z0, 3, 170, (int)SN3);
  // bilinear + global max -> out [B, 2046]
  final_max_kernel<<<B * 682, 256, 0, stream>>>(A0, z0, (float*)d_out);
}
